// MIModel_61692910240324
// MI455X (gfx1250) — matrix operand order repaired and checked by reading, not compiled, not run on hardware
//
#include <hip/hip_runtime.h>
#include <math.h>

typedef __attribute__((ext_vector_type(16))) _Float16 v16h;
typedef __attribute__((ext_vector_type(8)))  _Float16 v8h;
typedef __attribute__((ext_vector_type(2)))  _Float16 v2h;
typedef __attribute__((ext_vector_type(8)))  float    v8f;

#define BSZ 512
#define HS  136            // padded LDS row stride (halves): 272B, conflict-free A reloads
#define TST (16 * HS)      // one 16-row tile in halves
#define NWAVES 8

// LDS weight-arena offsets (halves); fragments stored as two 512B half-chunks
#define W2_OFF 0           // 128x128 -> 16384 halves
#define W3_OFF 16384       // 128x64  ->  8192 halves
#define W4_OFF 24576       //  64x64  ->  4096 halves
#define W5_OFF 28672       //  64x64  ->  4096 halves
#define WTOT   32768       // 64KB total

// ---------------------------------------------------------------- utilities
__global__ void zero_ws(float* p, int n) {
    for (int idx = threadIdx.x; idx < n; idx += blockDim.x) p[idx] = 0.0f;
}

// xW1[r] = x[r] @ W1[0:64,:]   ;   yW1[r] = y[r] @ W1[64:128,:] + b1
__global__ void precompute_xy(const float* __restrict__ x, const float* __restrict__ y,
                              const float* __restrict__ W1, const float* __restrict__ b1,
                              _Float16* __restrict__ xW1, _Float16* __restrict__ yW1) {
    int r = blockIdx.x;          // 0..1023 : first 512 = x rows, next 512 = y rows
    int c = threadIdx.x;         // 0..127
    if (r < BSZ) {
        float s = 0.0f;
        #pragma unroll 8
        for (int k = 0; k < 64; ++k) s += x[r * 64 + k] * W1[k * 128 + c];
        xW1[r * 128 + c] = (_Float16)s;
    } else {
        int rr = r - BSZ;
        float s = b1[c];
        #pragma unroll 8
        for (int k = 0; k < 64; ++k) s += y[rr * 64 + k] * W1[(64 + k) * 128 + c];
        yW1[rr * 128 + c] = (_Float16)s;
    }
}

// Repack row-major (K,N) fp32 weight into WMMA B-fragment order (contiguous k, same order as the A loads):
// lane L holds n = nt*16 + L%16 ; slot s holds k = kc*32 + s + 16*(L/16).
// Fragment storage = two 512B half-chunks (lane stride 16B) for 2-phase LDS b128 reads.
__global__ void repack_w(const float* __restrict__ src, _Float16* __restrict__ dst,
                         int K, int N) {
    int gid = blockIdx.x * 256 + threadIdx.x;
    if (gid >= K * N) return;
    int frag = gid >> 9, rem = gid & 511;
    int lane = rem >> 4, slot = rem & 15;
    int ntc = N >> 4;
    int kc = frag / ntc, nt = frag % ntc;
    int half = lane >> 4;
    int n = nt * 16 + (lane & 15);
    int k = kc * 32 + slot + 16 * half;
    dst[frag * 512 + ((slot >> 3) << 8) + (lane << 3) + (slot & 7)] =
        (_Float16)src[k * N + n];
}

__device__ __forceinline__ v16h load_bfrag(const _Float16* wlds, int frag, int lane) {
    v8h lo = *(const v8h*)(wlds + (frag << 9) + (lane << 3));
    v8h hi = *(const v8h*)(wlds + (frag << 9) + 256 + (lane << 3));
    return __builtin_shufflevector(lo, hi, 0, 1, 2, 3, 4, 5, 6, 7,
                                   8, 9, 10, 11, 12, 13, 14, 15);
}

// ---------------------------------------------------------------- fused g-MLP layer
// Two 16-row A tiles per wave (register M-blocking) and double-buffered B fragments:
// nt+1's fragments load from LDS while nt's WMMAs execute (partial dscnt waits).
template <int KC, int NT>
__device__ __forceinline__ void g_layer(_Float16* hb, const _Float16* wlds,
                                        const float* __restrict__ bias, int lane) {
    const int m = lane & 15, half = lane >> 4;
    const int col = lane & 15;
    v16h a0[KC], a1[KC];
    #pragma unroll
    for (int kc = 0; kc < KC; ++kc) {
        #pragma unroll
        for (int v = 0; v < 8; ++v) {
            // A 16x32 contiguous: V0..7 -> K = 2v + 16*half (matches B fragment order)
            int kb = kc * 32 + half * 16 + v * 2;
            v2h p0 = *(const v2h*)(hb + m * HS + kb);
            v2h p1 = *(const v2h*)(hb + TST + m * HS + kb);
            a0[kc][2 * v] = p0.x; a0[kc][2 * v + 1] = p0.y;
            a1[kc][2 * v] = p1.x; a1[kc][2 * v + 1] = p1.y;
        }
    }
    float bv[NT];
    #pragma unroll
    for (int nt = 0; nt < NT; ++nt) bv[nt] = bias[nt * 16 + col];
    __syncthreads();

    v16h bf[KC];
    #pragma unroll
    for (int kc = 0; kc < KC; ++kc) bf[kc] = load_bfrag(wlds, kc * NT, lane);

    #pragma unroll
    for (int nt = 0; nt < NT; ++nt) {
        v16h bn[KC];
        if (nt + 1 < NT) {
            #pragma unroll
            for (int kc = 0; kc < KC; ++kc)
                bn[kc] = load_bfrag(wlds, kc * NT + nt + 1, lane);
        }
        v8f acc0 = {}, acc1 = {};
        #pragma unroll
        for (int kc = 0; kc < KC; ++kc) {
            acc0 = __builtin_amdgcn_wmma_f32_16x16x32_f16(
                false, a0[kc], false, bf[kc], (short)0, acc0, false, false);
            acc1 = __builtin_amdgcn_wmma_f32_16x16x32_f16(
                false, a1[kc], false, bf[kc], (short)0, acc1, false, false);
        }
        #pragma unroll
        for (int r = 0; r < 8; ++r) {      // D layout: m = r + 8*half, n = col
            float v0 = fmaxf(acc0[r] + bv[nt], 0.0f);
            float v1 = fmaxf(acc1[r] + bv[nt], 0.0f);
            hb[(r + 8 * half) * HS + nt * 16 + col] = (_Float16)v0;
            hb[TST + (r + 8 * half) * HS + nt * 16 + col] = (_Float16)v1;
        }
        if (nt + 1 < NT) {
            #pragma unroll
            for (int kc = 0; kc < KC; ++kc) bf[kc] = bn[kc];
        }
    }
    __syncthreads();
}

__global__ void __launch_bounds__(256)
g_pairs_kernel(const _Float16* __restrict__ xW1, const _Float16* __restrict__ yW1,
               const _Float16* __restrict__ wall,   // w2r..w5r contiguous, 32768 halves
               const float* __restrict__ b2, const float* __restrict__ b3,
               const float* __restrict__ b4, const float* __restrict__ b5,
               const float* __restrict__ w6, const float* __restrict__ b6,
               float* __restrict__ rowsum, float* __restrict__ diagv) {
    __shared__ _Float16 wsh[WTOT];                 // 64KB staged weights, block-shared
    __shared__ _Float16 hbuf[NWAVES][2 * TST];     // 2 tiles x 16x136 per wave
    __shared__ float red[NWAVES][32];
    const int lane = threadIdx.x & 31;
    const int wv = threadIdx.x >> 5;
    const int t = blockIdx.x * NWAVES + wv;        // 8192 waves: 512 rows x 16 j-chunks
    const int i = t >> 4;                          // score row (uses x[i])
    const int j0 = (t & 15) << 5;                  // 32 columns (uses y[j0..j0+31])
    _Float16* hb = hbuf[wv];

    // Stage all weights into LDS once per block (amortized over 8 waves x 32 pairs)
    {
        const uint4* src = (const uint4*)wall;
        uint4* dst = (uint4*)wsh;
        for (int idx = threadIdx.x; idx < WTOT / 8; idx += 256) dst[idx] = src[idx];
    }

    // Stage h1[m][:] = relu(xW1[i] + yW1[j0+m])  (b1 folded into yW1), 32 rows
    for (int u = lane; u < 512; u += 32) {         // 32 rows x 16 chunks of 8 halves
        int m = u >> 4, c8 = (u & 15) << 3;
        v8h xa = *(const v8h*)(xW1 + i * 128 + c8);
        v8h yb = *(const v8h*)(yW1 + (j0 + m) * 128 + c8);
        v8h s = xa + yb;
        #pragma unroll
        for (int e = 0; e < 8; ++e)
            s[e] = s[e] > (_Float16)0.0f ? s[e] : (_Float16)0.0f;
        *(v8h*)(hb + (m >> 4) * TST + (m & 15) * HS + c8) = s;
    }
    __syncthreads();

    g_layer<4, 8>(hb, wsh + W2_OFF, b2, lane);   // 128 -> 128
    g_layer<4, 4>(hb, wsh + W3_OFF, b3, lane);   // 128 -> 64
    g_layer<2, 4>(hb, wsh + W4_OFF, b4, lane);   //  64 -> 64
    g_layer<2, 4>(hb, wsh + W5_OFF, b5, lane);   //  64 -> 64

    // head: 64 -> 1 (VALU dot over both tiles), then diagonal / off-diagonal reduce
    {
        float s = b6[0];
        const _Float16* hrow = hb + (lane >> 4) * TST + (lane & 15) * HS;
        #pragma unroll 8
        for (int c = 0; c < 64; ++c) s += (float)hrow[c] * w6[c];
        red[wv][lane] = s;
    }
    __syncthreads();
    if (lane == 0) {
        float acc = 0.0f, dg = 0.0f;
        #pragma unroll
        for (int m = 0; m < 32; ++m) {
            float v = red[wv][m];
            if (j0 + m == i) dg = v; else acc += v;
        }
        atomicAdd(&rowsum[i], acc);
        if (i >= j0 && i < j0 + 32) diagv[i] = dg;   // exactly one wave owns diag(i)
    }
}

// ---------------------------------------------------------------- alpha MLP + MI finalize
__global__ void __launch_bounds__(256)
alpha_finalize(const float* __restrict__ y,
               const float* __restrict__ aW1, const float* __restrict__ ab1,
               const float* __restrict__ aW2, const float* __restrict__ ab2,
               const float* __restrict__ aW3, const float* __restrict__ ab3,
               const float* __restrict__ rowsum, const float* __restrict__ diagv,
               float* __restrict__ out) {
    __shared__ float h1s[NWAVES][168];
    __shared__ float h2s[NWAVES][64];
    const int lane = threadIdx.x & 31;
    const int wv = threadIdx.x >> 5;
    const int row = blockIdx.x * NWAVES + wv;   // 512 rows, 64 blocks
    for (int c = lane; c < 168; c += 32) {
        float s = ab1[c];
        #pragma unroll 8
        for (int k = 0; k < 64; ++k) s += y[row * 64 + k] * aW1[k * 168 + c];
        h1s[wv][c] = fmaxf(s, 0.0f);
    }
    __syncthreads();
    for (int c = lane; c < 64; c += 32) {
        float s = ab2[c];
        for (int k = 0; k < 168; ++k) s += h1s[wv][k] * aW2[k * 64 + c];
        h2s[wv][c] = fmaxf(s, 0.0f);
    }
    __syncthreads();
    if (lane == 0) {
        float la = ab3[0];
        #pragma unroll 8
        for (int k = 0; k < 64; ++k) la += h2s[wv][k] * aW3[k];
        float joint = diagv[row] - la;
        float marg = rowsum[row] * (1.0f / 511.0f) - la;
        out[row] = 1.0f + joint - expf(marg);
    }
}

// ---------------------------------------------------------------- launch
extern "C" void kernel_launch(void* const* d_in, const int* in_sizes, int n_in,
                              void* d_out, int out_size, void* d_ws, size_t ws_size,
                              hipStream_t stream) {
    (void)in_sizes; (void)n_in; (void)out_size; (void)ws_size;
    const float* x   = (const float*)d_in[0];
    const float* yv  = (const float*)d_in[1];
    const float* gW1 = (const float*)d_in[2];  const float* gb1 = (const float*)d_in[3];
    const float* gW2 = (const float*)d_in[4];  const float* gb2 = (const float*)d_in[5];
    const float* gW3 = (const float*)d_in[6];  const float* gb3 = (const float*)d_in[7];
    const float* gW4 = (const float*)d_in[8];  const float* gb4 = (const float*)d_in[9];
    const float* gW5 = (const float*)d_in[10]; const float* gb5 = (const float*)d_in[11];
    const float* gW6 = (const float*)d_in[12]; const float* gb6 = (const float*)d_in[13];
    const float* aW1 = (const float*)d_in[14]; const float* ab1 = (const float*)d_in[15];
    const float* aW2 = (const float*)d_in[16]; const float* ab2 = (const float*)d_in[17];
    const float* aW3 = (const float*)d_in[18]; const float* ab3 = (const float*)d_in[19];

    char* ws = (char*)d_ws;
    float*    rowsum = (float*)(ws + 0);                 // 512 f32
    float*    diagv  = (float*)(ws + 2048);              // 512 f32
    _Float16* xW1    = (_Float16*)(ws + 4096);           // 512x128 f16
    _Float16* yW1    = (_Float16*)(ws + 4096 + 131072);  // 512x128 f16
    _Float16* wall   = (_Float16*)(ws + 4096 + 262144);  // repacked weights, contiguous
    _Float16* w2r    = wall;                             // 128x128
    _Float16* w3r    = wall + W3_OFF;                    // 128x64
    _Float16* w4r    = wall + W4_OFF;                    // 64x64
    _Float16* w5r    = wall + W5_OFF;                    // 64x64

    zero_ws<<<1, 256, 0, stream>>>(rowsum, 1024);        // rowsum + diag contiguous
    precompute_xy<<<1024, 128, 0, stream>>>(x, yv, gW1, gb1, xW1, yW1);
    repack_w<<<64, 256, 0, stream>>>(gW2, w2r, 128, 128);
    repack_w<<<32, 256, 0, stream>>>(gW3, w3r, 128, 64);
    repack_w<<<16, 256, 0, stream>>>(gW4, w4r, 64, 64);
    repack_w<<<16, 256, 0, stream>>>(gW5, w5r, 64, 64);
    g_pairs_kernel<<<1024, 256, 0, stream>>>(xW1, yW1, wall,
                                             gb2, gb3, gb4, gb5, gW6, gb6,
                                             rowsum, diagv);
    alpha_finalize<<<64, 256, 0, stream>>>(yv, aW1, ab1, aW2, ab2, aW3, ab3,
                                           rowsum, diagv, (float*)d_out);
}
